// Retriever_41257455845660
// MI455X (gfx1250) — matrix operand order repaired and checked by reading, not compiled, not run on hardware
//
#include <hip/hip_runtime.h>
#include <hip/hip_bf16.h>

typedef __attribute__((ext_vector_type(16))) __bf16 v16bf;
typedef __attribute__((ext_vector_type(8)))  __bf16 v8bf;
typedef __attribute__((ext_vector_type(4)))  __bf16 v4bf;
typedef __attribute__((ext_vector_type(8)))  float  v8f;
typedef __attribute__((ext_vector_type(4)))  float  v4f;

#define BQ     256          // queries
#define DIM    768
#define NEMB   500000
#define KOUT   17           // k+1
#define NB     512          // phase-1 blocks
#define NT     64           // embeddings per group
#define KC     64           // k-chunk (LDS staging)
#define KCP    72           // padded LDS row stride in bf16 (bank-conflict-free)
#define KSTEPS (DIM / 32)   // 24 WMMA k-steps over D
#define NCHUNK (DIM / KC)   // 12

// ---------------- Phase 0a: query inverse norms ----------------
__global__ __launch_bounds__(256) void qnorm_kernel(const float* __restrict__ q,
                                                    float* __restrict__ rinv) {
    __shared__ float red[256];
    int m = blockIdx.x;
    float s = 0.f;
    for (int d = threadIdx.x; d < DIM; d += 256) {
        float v = q[m * DIM + d];
        s += v * v;
    }
    red[threadIdx.x] = s;
    __syncthreads();
    for (int st = 128; st > 0; st >>= 1) {
        if (threadIdx.x < st) red[threadIdx.x] += red[threadIdx.x + st];
        __syncthreads();
    }
    if (threadIdx.x == 0) rinv[m] = rsqrtf(red[0]);
}

// ---------------- Phase 0b: pack normalized queries into WMMA A-fragment layout -------
// A-fragment (16-bit, 16x32): lane<16 : M=lane,    halfword j -> K = (j<8)? j   : j+8
//                             lane>=16: M=lane-16, halfword j -> K = (j<8)? j+8 : j+16
__global__ __launch_bounds__(256) void qpack_kernel(const float* __restrict__ q,
                                                    const float* __restrict__ rinv,
                                                    __bf16* __restrict__ qpack) {
    int mt = blockIdx.x / KSTEPS;
    int ks = blockIdx.x % KSTEPS;
    for (int h = threadIdx.x; h < 512; h += 256) {
        int lane = h >> 4, j = h & 15;
        int m = mt * 16 + (lane & 15);
        int kl = (lane < 16) ? ((j < 8) ? j : (j + 8))
                             : ((j < 8) ? (j + 8) : (j + 16));
        int k = ks * 32 + kl;
        float v = q[m * DIM + k] * rinv[m];
        qpack[(((size_t)mt * KSTEPS + ks) * 32 + lane) * 16 + j] = (__bf16)v;
    }
}

// ---------------- Phase 1: streamed WMMA GEMM + fused norm + block-local top-17 -------
__global__ __attribute__((amdgpu_num_vgpr(256))) __launch_bounds__(256)
void sim_topk_kernel(const float*  __restrict__ emb,
                     const __bf16* __restrict__ qpack,
                     float* __restrict__ pv,
                     int*   __restrict__ pi) {
    __shared__ __bf16 eb16[2][NT * KCP];   // 2 x 9 KB bf16 chunk, double-buffered
    __shared__ float topv[BQ * KOUT];      // 17 KB
    __shared__ int   topi[BQ * KOUT];      // 17 KB
    __shared__ float rowmin[BQ];
    __shared__ int   minpos[BQ];
    __shared__ float ssq[NT];
    __shared__ float rnq[NT];

    const int tid  = threadIdx.x;
    const int lane = tid & 31;
    const int w    = tid >> 5;          // wave id 0..7
    const int le   = tid >> 2;          // embedding row this thread stages (0..63)
    const int lp   = tid & 3;           // part within row
    const int ln15 = lane & 15;
    const int hk   = (lane & 16) ? 8 : 0;    // B-fragment K offset for upper half-wave

    // init per-block top lists
    {
        int q = tid;
        for (int s = 0; s < KOUT; ++s) { topv[q * KOUT + s] = -INFINITY; topi[q * KOUT + s] = 0; }
        rowmin[q] = -INFINITY;
        minpos[q] = 0;
    }
    __syncthreads();

    const int ngroups = (NEMB + NT - 1) / NT;   // 7813

    for (int g = blockIdx.x; g < ngroups; g += NB) {
        v8f acc[2][4];
        #pragma unroll
        for (int a = 0; a < 2; ++a)
            #pragma unroll
            for (int b = 0; b < 4; ++b) acc[a][b] = v8f{};

        float myssq = 0.f;
        const int eg = g * NT + le;
        const bool valid = (eg < NEMB);
        const float* src = emb + (size_t)eg * DIM;

        // preload chunk 0 into registers
        v4f ld[4];
        #pragma unroll
        for (int i = 0; i < 4; ++i)
            ld[i] = valid ? *(const v4f*)(src + (i * 4 + lp) * 4) : v4f{};

        int buf = 0;
        #pragma unroll 1
        for (int kc = 0; kc < NCHUNK; ++kc) {
            // ---- stage registers -> LDS (f32 -> bf16 once, here only) ----
            #pragma unroll
            for (int i = 0; i < 4; ++i) {
                v4f v = ld[i];
                myssq += v.x * v.x + v.y * v.y + v.z * v.z + v.w * v.w;
                v4bf h;
                h[0] = (__bf16)v.x; h[1] = (__bf16)v.y;
                h[2] = (__bf16)v.z; h[3] = (__bf16)v.w;
                *(v4bf*)&eb16[buf][le * KCP + (i * 4 + lp) * 4] = h;
            }
            __syncthreads();   // single barrier per chunk (safe with ping-pong buffers)

            // ---- issue next chunk's HBM loads; they overlap the WMMAs below ----
            if (kc + 1 < NCHUNK && valid) {
                #pragma unroll
                for (int i = 0; i < 4; ++i)
                    ld[i] = *(const v4f*)(src + (kc + 1) * KC + (i * 4 + lp) * 4);
                if (kc + 2 < NCHUNK)
                    __builtin_prefetch(src + (kc + 2) * KC + lp * 16, 0, 0);
            }

            // ---- compute on current buffer: 2 k-steps x (2 M x 4 N) WMMAs ----
            #pragma unroll
            for (int ks = 0; ks < KC / 32; ++ks) {
                int kstep = kc * (KC / 32) + ks;
                v16bf afrag[2];
                #pragma unroll
                for (int mi = 0; mi < 2; ++mi) {
                    int mt = w * 2 + mi;
                    afrag[mi] = *(const v16bf*)(qpack +
                        (((size_t)mt * KSTEPS + kstep) * 32 + lane) * 16);
                }
                #pragma unroll
                for (int nt = 0; nt < 4; ++nt) {
                    // B-fragment: lane<16 -> N=lane, K=0..7,16..23 ; lane>=16 -> N=lane-16, K=8..15,24..31
                    const __bf16* bp = &eb16[buf][(nt * 16 + ln15) * KCP + ks * 32 + hk];
                    v8bf lo = *(const v8bf*)bp;
                    v8bf hi = *(const v8bf*)(bp + 16);
                    v16bf bfrag = __builtin_shufflevector(lo, hi,
                        0, 1, 2, 3, 4, 5, 6, 7, 8, 9, 10, 11, 12, 13, 14, 15);
                    #pragma unroll
                    for (int mi = 0; mi < 2; ++mi) {
                        acc[mi][nt] = __builtin_amdgcn_wmma_f32_16x16x32_bf16(
                            false, afrag[mi], false, bfrag,
                            (short)0, acc[mi][nt], false, false);
                    }
                }
            }
            buf ^= 1;
        }

        // ---- reduce embedding sum-of-squares -> inverse norms ----
        __syncthreads();
        if (tid < NT) ssq[tid] = 0.f;
        __syncthreads();
        atomicAdd(&ssq[le], myssq);
        __syncthreads();
        if (tid < NT) rnq[tid] = rsqrtf(ssq[tid]);
        __syncthreads();

        // ---- top-k update: wave w exclusively owns queries [w*32, w*32+32) ----
        #pragma unroll
        for (int mi = 0; mi < 2; ++mi) {
            int mtile = w * 2 + mi;
            #pragma unroll
            for (int r = 0; r < 8; ++r) {
                int q = mtile * 16 + ((lane & 16) ? 8 : 0) + r;
                float rmin = rowmin[q];   // stale-low is safe; re-checked under leader
                #pragma unroll
                for (int nt = 0; nt < 4; ++nt) {
                    int el = nt * 16 + ln15;
                    int gi = g * NT + el;
                    float sim = (gi < NEMB) ? acc[mi][nt][r] * rnq[el] : -INFINITY;
                    unsigned long long mask = __ballot(sim > rmin);
                    while (mask) {
                        int leader = __ffsll(mask) - 1;
                        if (lane == leader) {
                            float rm = *(volatile float*)&rowmin[q];
                            if (sim > rm) {
                                int mp = minpos[q];
                                topv[q * KOUT + mp] = sim;
                                topi[q * KOUT + mp] = gi;
                                float nm = topv[q * KOUT];
                                int   np = 0;
                                #pragma unroll
                                for (int s = 1; s < KOUT; ++s) {
                                    float tv = topv[q * KOUT + s];
                                    if (tv < nm) { nm = tv; np = s; }
                                }
                                rowmin[q] = nm;
                                minpos[q] = np;
                            }
                        }
                        mask &= mask - 1;
                    }
                }
            }
        }
        __syncthreads();
    }

    __syncthreads();
    // dump block-local lists to workspace
    {
        int q = tid;
        size_t base = ((size_t)blockIdx.x * BQ + q) * KOUT;
        for (int s = 0; s < KOUT; ++s) {
            pv[base + s] = topv[q * KOUT + s];
            pi[base + s] = topi[q * KOUT + s];
        }
    }
}

// ---------------- Phase 2: merge NB partial lists per query -> final top-17 ----------
__global__ __launch_bounds__(256) void merge_kernel(float* __restrict__ pv,
                                                    const int* __restrict__ pi,
                                                    float* __restrict__ ov,
                                                    int*   __restrict__ oi) {
    __shared__ float rv[256];
    __shared__ int   re[256];
    const int q = blockIdx.x, tid = threadIdx.x;
    const int total = NB * KOUT;   // 8704 candidates per query

    for (int round = 0; round < KOUT; ++round) {
        float best = -INFINITY;
        int   beste = 0;
        for (int e = tid; e < total; e += 256) {
            int b = e / KOUT, s = e % KOUT;
            float v = pv[((size_t)b * BQ + q) * KOUT + s];
            if (v > best) { best = v; beste = e; }
        }
        rv[tid] = best; re[tid] = beste;
        __syncthreads();
        for (int st = 128; st > 0; st >>= 1) {
            if (tid < st && rv[tid + st] > rv[tid]) {
                rv[tid] = rv[tid + st];
                re[tid] = re[tid + st];
            }
            __syncthreads();
        }
        if (tid == 0) {
            int e = re[0], b = e / KOUT, s = e % KOUT;
            size_t addr = ((size_t)b * BQ + q) * KOUT + s;
            ov[q * KOUT + round] = rv[0];
            oi[q * KOUT + round] = pi[addr];
            pv[addr] = -INFINITY;              // pop for next round
        }
        __syncthreads();
    }
}

// ---------------- host-side launcher ----------------
extern "C" void kernel_launch(void* const* d_in, const int* in_sizes, int n_in,
                              void* d_out, int out_size, void* d_ws, size_t ws_size,
                              hipStream_t stream) {
    const float* q   = (const float*)d_in[0];
    const float* emb = (const float*)d_in[1];
    // d_in[2] is k==16 -> KOUT = 17 baked in

    char* ws = (char*)d_ws;
    float*  rinv  = (float*)ws;                                              // 1 KB
    __bf16* qpack = (__bf16*)(ws + 1024);                                    // 384 KB
    size_t  qpack_bytes = (size_t)16 * KSTEPS * 32 * 16 * sizeof(__bf16);
    float*  pv = (float*)(ws + 1024 + qpack_bytes);                          // 8.9 MB
    int*    pi = (int*)((char*)pv + (size_t)NB * BQ * KOUT * sizeof(float)); // 8.9 MB

    float* ov = (float*)d_out;
    int*   oi = (int*)((float*)d_out + (size_t)BQ * KOUT);

    qnorm_kernel   <<<BQ,          256, 0, stream>>>(q, rinv);
    qpack_kernel   <<<16 * KSTEPS, 256, 0, stream>>>(q, rinv, qpack);
    sim_topk_kernel<<<NB,          256, 0, stream>>>(emb, qpack, pv, pi);
    merge_kernel   <<<BQ,          256, 0, stream>>>(pv, pi, ov, oi);
}
